// WindowAttention_32710470927072
// MI455X (gfx1250) — matrix operand order repaired and checked by reading, not compiled, not run on hardware
//
#include <hip/hip_runtime.h>
#include <hip/hip_bf16.h>

// ---- problem constants -----------------------------------------------------
#define WWH   7
#define NWIN  49          // tokens per window
#define NPAD  64          // padded to 4x 16-row tiles
#define HEADS 4
#define HD    32          // head dim
#define CDIM  128
#define TBL   169         // (2*7-1)^2 bias table rows
#define SCALE 0.17677669529663687f   // 32^-0.5

typedef __attribute__((ext_vector_type(16))) __bf16 v16bf;
typedef __attribute__((ext_vector_type(8)))  float  v8f;

union Frag16 { v16bf v; uint4 q[2]; };
union FragF  { v8f  v; float f[8]; };
union Pack4  { __bf16 h[4]; uint2 u; };

// A-matrix 16x32 bf16 fragment: lane L -> row (L&15), K chunks
// [ (L>>4)*8 .. +8 ) and [ 16+(L>>4)*8 .. +8 )   (two 16B loads)
static __device__ __forceinline__ void load_a(const __bf16* base, int stride,
                                              int row, int k0, int lane, Frag16& fr) {
  const __bf16* p = base + row * stride + k0 + ((lane >> 4) << 3);
  fr.q[0] = *reinterpret_cast<const uint4*>(p);
  fr.q[1] = *reinterpret_cast<const uint4*>(p + 16);
}

// B-matrix 32x16 bf16 fragment: lane L -> column (L&15), K chunks
// [ (L>>4)*8 .. +8 ) and [ 16+(L>>4)*8 .. +8 ). colbase points at start of that column's K-run.
static __device__ __forceinline__ void load_b(const __bf16* colbase,
                                              int k0, int lane, Frag16& fr) {
  const __bf16* p = colbase + k0 + ((lane >> 4) << 3);
  fr.q[0] = *reinterpret_cast<const uint4*>(p);
  fr.q[1] = *reinterpret_cast<const uint4*>(p + 16);
}

static __device__ __forceinline__ v8f wmma_bf16(const Frag16& a, const Frag16& b, v8f c) {
  return __builtin_amdgcn_wmma_f32_16x16x32_bf16(false, a.v, false, b.v,
                                                 (short)0, c, false, false);
}

// 4-wmma K=128 column-tile GEMM against a bf16 weight matrix (row-major [col][128])
static __device__ __forceinline__ v8f gemm_k128(const Frag16 a[4], const __bf16* wcol,
                                                int lane) {
  FragF acc; acc.v = v8f{0,0,0,0,0,0,0,0};
#pragma unroll
  for (int c = 0; c < 4; ++c) {
    Frag16 b; load_b(wcol, 32 * c, lane, b);
    acc.v = wmma_bf16(a[c], b, acc.v);
  }
  return acc.v;
}

// ---- weight fp32 -> bf16 conversion (weights then live in L2) --------------
__global__ void cvt_weights_kernel(const float* __restrict__ qkvw,
                                   const float* __restrict__ projw,
                                   __bf16* __restrict__ out_qkvw,
                                   __bf16* __restrict__ out_projw) {
  int i = blockIdx.x * 256 + threadIdx.x;
  if (i < 3 * CDIM * CDIM) out_qkvw[i] = (__bf16)qkvw[i];
  if (i < CDIM * CDIM)     out_projw[i] = (__bf16)projw[i];
}

// ---- fused window attention: one workgroup per window ----------------------
// LDS layout (bytes):
//   sx    [64][128] bf16   @      0  (16KB)
//   sq    [4][64][32] bf16 @  16384  (16KB)   pre-scaled q
//   sk    [4][64][32] bf16 @  32768  (16KB)
//   svT   [4][32][64] bf16 @  49152  (16KB)   v transposed (d-major)
//   sP    [4][64][64] bf16 @  65536  (32KB)   softmax probs
//   so    [64][128] bf16   @  98304  (16KB)   attention output (pre-proj)
//   sbias [169][4] f32     @ 114688  (2704B)  relative position bias table
#define SMEM_BYTES (114688 + 2704)

__global__ void __launch_bounds__(256, 2)
window_attn_kernel(const float* __restrict__ x,
                   const float* __restrict__ qkv_b,
                   const float* __restrict__ proj_b,
                   const float* __restrict__ bias_table,
                   const __bf16* __restrict__ wqkv,   // [384][128] bf16
                   const __bf16* __restrict__ wproj,  // [128][128] bf16
                   float* __restrict__ out) {
  extern __shared__ __align__(16) char smem[];
  __bf16* sx    = reinterpret_cast<__bf16*>(smem);
  __bf16* sq    = reinterpret_cast<__bf16*>(smem + 16384);
  __bf16* sk    = reinterpret_cast<__bf16*>(smem + 32768);
  __bf16* svT   = reinterpret_cast<__bf16*>(smem + 49152);
  __bf16* sP    = reinterpret_cast<__bf16*>(smem + 65536);
  __bf16* so    = reinterpret_cast<__bf16*>(smem + 98304);
  float*  sbias = reinterpret_cast<float*>(smem + 114688);

  const int tid  = threadIdx.x;
  const int w    = tid >> 5;      // wave id 0..7
  const int lane = tid & 31;
  const int lo   = lane & 15;
  const int hi   = lane >> 4;
  const long long win = blockIdx.x;
  const float* xw = x + win * (NWIN * CDIM);

  // ---- stage 0: bias table -> LDS; x -> LDS bf16 (vectorized) --------------
  for (int i = tid; i < TBL * HEADS; i += 256) sbias[i] = bias_table[i];
#pragma unroll
  for (int c = tid; c < (NPAD * CDIM) / 4; c += 256) {   // 8 float4 per thread
    int row = c >> 5, col4 = (c & 31) << 2;
    float4 v = make_float4(0.f, 0.f, 0.f, 0.f);
    if (row < NWIN) v = *reinterpret_cast<const float4*>(xw + row * CDIM + col4);
    Pack4 p;
    p.h[0] = (__bf16)v.x; p.h[1] = (__bf16)v.y;
    p.h[2] = (__bf16)v.z; p.h[3] = (__bf16)v.w;
    *reinterpret_cast<uint2*>(sx + row * CDIM + col4) = p.u;
  }
  __syncthreads();

  // ---- stage 1: qkv = x @ Wqkv^T + b  (64x384, tiles 16x16, K=128) ---------
  // 24 col-tiles: [0,8)=q [8,16)=k [16,24)=v. Wave (mt = w&3, g = w>>2)
  // handles 4 tiles per section -> compile-time store paths, no branches.
  {
    const int mt = w & 3;
    const int g  = w >> 2;
    Frag16 a[4];
#pragma unroll
    for (int c = 0; c < 4; ++c)
      load_a(sx, CDIM, mt * 16 + lo, 32 * c, lane, a[c]);

    // ---- Q tiles (pre-scaled) ----
#pragma unroll
    for (int t = 0; t < 4; ++t) {
      const int nt  = g * 4 + t;                     // 0..7
      const int col = nt * 16 + lo;
      FragF acc; acc.v = gemm_k128(a, wqkv + col * CDIM, lane);
      const float bias = qkv_b[col];
      const int hh = nt >> 1, d = ((nt & 1) << 4) + lo;
#pragma unroll
      for (int r = 0; r < 8; ++r) {
        int m = mt * 16 + r + (hi << 3);
        sq[(hh * NPAD + m) * HD + d] = (__bf16)((acc.f[r] + bias) * SCALE);
      }
    }
    // ---- K tiles ----
#pragma unroll
    for (int t = 0; t < 4; ++t) {
      const int nt  = 8 + g * 4 + t;                 // 8..15
      const int col = nt * 16 + lo;
      FragF acc; acc.v = gemm_k128(a, wqkv + col * CDIM, lane);
      const float bias = qkv_b[col];
      const int hh = (nt >> 1) & 3, d = ((nt & 1) << 4) + lo;
#pragma unroll
      for (int r = 0; r < 8; ++r) {
        int m = mt * 16 + r + (hi << 3);
        sk[(hh * NPAD + m) * HD + d] = (__bf16)(acc.f[r] + bias);
      }
    }
    // ---- V tiles (stored transposed: [h][d][m]) ----
#pragma unroll
    for (int t = 0; t < 4; ++t) {
      const int nt  = 16 + g * 4 + t;                // 16..23
      const int col = nt * 16 + lo;
      FragF acc; acc.v = gemm_k128(a, wqkv + col * CDIM, lane);
      const float bias = qkv_b[col];
      const int hh = (nt >> 1) & 3, d = ((nt & 1) << 4) + lo;
#pragma unroll
      for (int r = 0; r < 8; ++r) {
        int m = mt * 16 + r + (hi << 3);
        svT[(hh * HD + d) * NPAD + m] = (__bf16)(acc.f[r] + bias);
      }
    }
  }
  __syncthreads();

  // ---- stage 2: S = q k^T + relbias ; softmax ; P -> LDS bf16 --------------
  const int h    = w >> 1;              // 2 waves per head
  const int half = w & 1;               // rows [32*half, 32*half+32)
  const __bf16* sqh  = sq  + h * NPAD * HD;
  const __bf16* skh  = sk  + h * NPAD * HD;
  const __bf16* svTh = svT + h * HD * NPAD;
  __bf16*       sPh  = sP  + h * NPAD * NPAD;
  {
    FragF s[2][4];
#pragma unroll
    for (int mi = 0; mi < 2; ++mi) {
      const int mt = half * 2 + mi;
      Frag16 aq; load_a(sqh, HD, mt * 16 + lo, 0, lane, aq);
#pragma unroll
      for (int nt = 0; nt < 4; ++nt) {
        Frag16 bk; load_b(skh + (nt * 16 + lo) * HD, 0, lane, bk);
        s[mi][nt].v = wmma_bf16(aq, bk, v8f{0,0,0,0,0,0,0,0});
      }
    }
    // bias + mask + softmax (rows live in 16-lane groups; wave32 shuffles)
#pragma unroll
    for (int mi = 0; mi < 2; ++mi) {
      const int mt = half * 2 + mi;
#pragma unroll
      for (int r = 0; r < 8; ++r) {
        const int n  = mt * 16 + r + (hi << 3);
        const int ni = n / WWH, nj = n - ni * WWH;
        float rmax = -3.0e38f;
#pragma unroll
        for (int nt = 0; nt < 4; ++nt) {
          const int m = nt * 16 + lo;
          float add;
          if (m < NWIN) {
            if (n < NWIN) {
              int mi7 = m / WWH, mj = m - mi7 * WWH;
              int idx = ((ni - mi7) + 6) * 13 + ((nj - mj) + 6);
              add = sbias[idx * HEADS + h];
            } else add = 0.0f;
          } else add = -1.0e30f;                    // mask padded keys
          float sv = s[mi][nt].f[r] + add;
          s[mi][nt].f[r] = sv;
          rmax = fmaxf(rmax, sv);
        }
        rmax = fmaxf(rmax, __shfl_xor(rmax, 1));
        rmax = fmaxf(rmax, __shfl_xor(rmax, 2));
        rmax = fmaxf(rmax, __shfl_xor(rmax, 4));
        rmax = fmaxf(rmax, __shfl_xor(rmax, 8));
        float sum = 0.0f;
#pragma unroll
        for (int nt = 0; nt < 4; ++nt) {
          float p = __expf(s[mi][nt].f[r] - rmax);
          s[mi][nt].f[r] = p;
          sum += p;
        }
        sum += __shfl_xor(sum, 1);
        sum += __shfl_xor(sum, 2);
        sum += __shfl_xor(sum, 4);
        sum += __shfl_xor(sum, 8);
        const float inv = 1.0f / sum;
#pragma unroll
        for (int nt = 0; nt < 4; ++nt)
          sPh[n * NPAD + nt * 16 + lo] = (__bf16)(s[mi][nt].f[r] * inv);
      }
    }
  }
  __syncthreads();

  // ---- stage 3: O = P @ v  (per head: 64x64 @ 64x32), into so[64][128] -----
  {
#pragma unroll
    for (int mi = 0; mi < 2; ++mi) {
      const int mt = half * 2 + mi;
      Frag16 ap[2];
#pragma unroll
      for (int c = 0; c < 2; ++c)
        load_a(sPh, NPAD, mt * 16 + lo, 32 * c, lane, ap[c]);
#pragma unroll
      for (int dt = 0; dt < 2; ++dt) {
        FragF acc; acc.v = v8f{0,0,0,0,0,0,0,0};
#pragma unroll
        for (int c = 0; c < 2; ++c) {
          Frag16 bv; load_b(svTh + (dt * 16 + lo) * NPAD, 32 * c, lane, bv);
          acc.v = wmma_bf16(ap[c], bv, acc.v);
        }
        const int col = h * HD + dt * 16 + lo;      // 0..127
#pragma unroll
        for (int r = 0; r < 8; ++r) {
          int n = mt * 16 + r + (hi << 3);
          so[n * CDIM + col] = (__bf16)acc.f[r];
        }
      }
    }
  }
  __syncthreads();

  // ---- stage 4: out = O @ Wproj^T + b, fp32 store --------------------------
  {
    const int mt  = w & 3;
    const int nt0 = (w >> 2) * 4;       // 4 of 8 col-tiles per wave
    Frag16 a[4];
#pragma unroll
    for (int c = 0; c < 4; ++c)
      load_a(so, CDIM, mt * 16 + lo, 32 * c, lane, a[c]);
    float* outw = out + win * (NWIN * CDIM);
#pragma unroll
    for (int t = 0; t < 4; ++t) {
      const int nt  = nt0 + t;
      const int col = nt * 16 + lo;
      FragF acc; acc.v = gemm_k128(a, wproj + col * CDIM, lane);
      const float bias = proj_b[col];
#pragma unroll
      for (int r = 0; r < 8; ++r) {
        int n = mt * 16 + r + (hi << 3);
        if (n < NWIN) outw[n * CDIM + col] = acc.f[r] + bias;
      }
    }
  }
}

// ---- launch ----------------------------------------------------------------
extern "C" void kernel_launch(void* const* d_in, const int* in_sizes, int n_in,
                              void* d_out, int out_size, void* d_ws, size_t ws_size,
                              hipStream_t stream) {
  const float* x     = (const float*)d_in[0];
  const float* qkvw  = (const float*)d_in[1];
  const float* qkvb  = (const float*)d_in[2];
  const float* projw = (const float*)d_in[3];
  const float* projb = (const float*)d_in[4];
  const float* btab  = (const float*)d_in[5];

  __bf16* ws_qkvw  = (__bf16*)d_ws;                     // 3*128*128 bf16
  __bf16* ws_projw = ws_qkvw + 3 * CDIM * CDIM;         // 128*128 bf16

  cvt_weights_kernel<<<(3 * CDIM * CDIM + 255) / 256, 256, 0, stream>>>(
      qkvw, projw, ws_qkvw, ws_projw);

  const int nwin = in_sizes[0] / (NWIN * CDIM);         // 4096
  window_attn_kernel<<<nwin, 256, SMEM_BYTES, stream>>>(
      x, qkvb, projb, btab, ws_qkvw, ws_projw, (float*)d_out);
}
